// TransformerBlock_35218731827894
// MI455X (gfx1250) — matrix operand order repaired and checked by reading, not compiled, not run on hardware
//
#include <hip/hip_runtime.h>
#include <hip/hip_bf16.h>
#include <math.h>

// ---------------- problem constants ----------------
#define BB   4
#define LL   2048
#define ANGC 16
#define AMPC 48
#define NH   4
#define AQK  12
#define GQK  6
#define AV   12
#define GV   6
#define HD   24        // per-head q/k/v dim (6+6+12)
#define DPAD 32        // padded head dim for WMMA K
#define CI   64
#define KK   4
#define GH   96        // GRU hidden (2*AMP)
#define TC   2051      // conv output length (2054 padded - 4 + 1)

typedef __attribute__((ext_vector_type(16))) _Float16 v16h;
typedef __attribute__((ext_vector_type(8)))  _Float16 v8h;
typedef __attribute__((ext_vector_type(8)))  float    v8f;
typedef __attribute__((ext_vector_type(4)))  int      v4i;

__device__ __forceinline__ float sigm(float x) { return 1.0f / (1.0f + __expf(-x)); }

// -------- gfx1250 async global->LDS staging (ASYNCcnt path), with fallback ----
#if __has_builtin(__builtin_amdgcn_global_load_async_to_lds_b128) && \
    __has_builtin(__builtin_amdgcn_s_wait_asynccnt)
#define HAVE_ASYNC_LDS 1
#else
#define HAVE_ASYNC_LDS 0
#endif

#if HAVE_ASYNC_LDS
typedef __attribute__((address_space(1))) v4i gv4i;   // global
typedef __attribute__((address_space(3))) v4i lv4i;   // LDS
#endif

__device__ __forceinline__ void copy16(const void* g, void* l) {
#if HAVE_ASYNC_LDS
    __builtin_amdgcn_global_load_async_to_lds_b128((gv4i*)g, (lv4i*)l, 0, 0);
#else
    *(uint4*)l = *(const uint4*)g;
#endif
}
__device__ __forceinline__ void copy_fence() {
#if HAVE_ASYNC_LDS
    __builtin_amdgcn_s_wait_asynccnt(0);
#endif
}

// ============================================================
// Kernel 1: norms + QKV projections, write f16 Q/K ([BH,L,32]) and V^T ([BH,32,L])
// ============================================================
__global__ __launch_bounds__(256) void k_qkv(
    const float* __restrict__ xang, const float* __restrict__ xamp,
    const float* __restrict__ g_ang1, const float* __restrict__ g_amp1,
    const float* __restrict__ wq_amp, const float* __restrict__ wk_amp,
    const float* __restrict__ wq_ar, const float* __restrict__ wq_ai,
    const float* __restrict__ wk_ar, const float* __restrict__ wk_ai,
    const float* __restrict__ wv_amp, const float* __restrict__ wv_ar,
    const float* __restrict__ wv_ai,
    _Float16* __restrict__ Qh, _Float16* __restrict__ Kh, _Float16* __restrict__ Vt)
{
    int t = blockIdx.x * blockDim.x + threadIdx.x;
    if (t >= BB * LL) return;
    int b = t / LL, l = t % LL;

    float r[ANGC], im[ANGC], na[AMPC];
    const float* xa = xang + (size_t)t * (2 * ANGC);
    float ss = 0.f;
#pragma unroll
    for (int j = 0; j < ANGC; ++j) { r[j] = xa[j]; im[j] = xa[ANGC + j]; ss += r[j]*r[j] + im[j]*im[j]; }
    float inv = rsqrtf(ss / ANGC + 1e-10f);
#pragma unroll
    for (int j = 0; j < ANGC; ++j) { float g = g_ang1[j] * inv; r[j] *= g; im[j] *= g; }

    const float* xm = xamp + (size_t)t * AMPC;
    ss = 0.f;
#pragma unroll
    for (int j = 0; j < AMPC; ++j) { na[j] = xm[j]; ss += na[j]*na[j]; }
    inv = rsqrtf(ss / AMPC + 1e-9f);
#pragma unroll
    for (int j = 0; j < AMPC; ++j) na[j] *= inv * g_amp1[j];

    const float qscale = 0.2041241452319315f; // 1/sqrt(24)

    for (int h = 0; h < NH; ++h) {
        size_t base = ((size_t)(b * NH + h) * LL + l) * DPAD;
        size_t vb   = ((size_t)(b * NH + h) * DPAD) * LL + l;
        float vv[HD];
        // ---- Q ----
        {
            float qv[HD];
#pragma unroll
            for (int o = 0; o < GQK; ++o) {
                int oo = h * GQK + o; float ar = 0.f, ai = 0.f;
#pragma unroll
                for (int j = 0; j < ANGC; ++j) {
                    float wr = wq_ar[oo*ANGC+j], wi = wq_ai[oo*ANGC+j];
                    ar += r[j]*wr - im[j]*wi;  ai += r[j]*wi + im[j]*wr;
                }
                qv[o] = ar; qv[GQK + o] = ai;
            }
#pragma unroll
            for (int o = 0; o < AQK; ++o) {
                int oo = h * AQK + o; float s = 0.f;
#pragma unroll
                for (int j = 0; j < AMPC; ++j) s += na[j] * wq_amp[oo*AMPC+j];
                qv[2*GQK + o] = s;
            }
#pragma unroll
            for (int d = 0; d < DPAD; ++d) Qh[base + d] = (_Float16)(d < HD ? qv[d] * qscale : 0.f);
        }
        // ---- K ----
        {
            float kv[HD];
#pragma unroll
            for (int o = 0; o < GQK; ++o) {
                int oo = h * GQK + o; float ar = 0.f, ai = 0.f;
#pragma unroll
                for (int j = 0; j < ANGC; ++j) {
                    float wr = wk_ar[oo*ANGC+j], wi = wk_ai[oo*ANGC+j];
                    ar += r[j]*wr - im[j]*wi;  ai += r[j]*wi + im[j]*wr;
                }
                kv[o] = ar; kv[GQK + o] = ai;
            }
#pragma unroll
            for (int o = 0; o < AQK; ++o) {
                int oo = h * AQK + o; float s = 0.f;
#pragma unroll
                for (int j = 0; j < AMPC; ++j) s += na[j] * wk_amp[oo*AMPC+j];
                kv[2*GQK + o] = s;
            }
#pragma unroll
            for (int d = 0; d < DPAD; ++d) Kh[base + d] = (_Float16)(d < HD ? kv[d] : 0.f);
        }
        // ---- V (stored transposed: [BH, 32, L]) ----
#pragma unroll
        for (int o = 0; o < GV; ++o) {
            int oo = h * GV + o; float ar = 0.f, ai = 0.f;
#pragma unroll
            for (int j = 0; j < ANGC; ++j) {
                float wr = wv_ar[oo*ANGC+j], wi = wv_ai[oo*ANGC+j];
                ar += r[j]*wr - im[j]*wi;  ai += r[j]*wi + im[j]*wr;
            }
            vv[o] = ar; vv[GV + o] = ai;
        }
#pragma unroll
        for (int o = 0; o < AV; ++o) {
            int oo = h * AV + o; float s = 0.f;
#pragma unroll
            for (int j = 0; j < AMPC; ++j) s += na[j] * wv_amp[oo*AMPC+j];
            vv[2*GV + o] = s;
        }
#pragma unroll
        for (int d = 0; d < DPAD; ++d) Vt[vb + (size_t)d * LL] = (_Float16)(d < HD ? vv[d] : 0.f);
    }
}

// ============================================================
// Kernel 2: flash attention, v_wmma_f32_16x16x32_f16 (wave32)
// block = 4 waves = 64 query rows of one (b,h); K/V chunks shared via
// double-buffered async global->LDS staging; shift-free softmax
// (scores are provably O(1): exp never overflows; max-subtraction is a
//  mathematical no-op for softmax) -> single epilogue reduction.
// ============================================================
__global__ __launch_bounds__(128) void k_flash(
    const _Float16* __restrict__ Qh, const _Float16* __restrict__ Kh,
    const _Float16* __restrict__ Vt, float* __restrict__ O)
{
    __shared__ __align__(32) _Float16 p_lds[4][16][32];   // per-wave P roundtrip
    __shared__ __align__(32) _Float16 Klds[2][32][32];    // 32 keys x 32 dims
    __shared__ __align__(32) _Float16 Vlds[2][32][32];    // 32 dims x 32 keys

    const int wave = threadIdx.x >> 5;
    const int lane = threadIdx.x & 31;
    const int l15  = lane & 15;
    const int hi   = lane >> 4;                  // 0: lanes 0-15, 1: lanes 16-31
    const int tiles64 = LL / 64;
    const int bh = blockIdx.x / tiles64;
    const int q0 = (blockIdx.x % tiles64) * 64 + wave * 16;
    const int c0 = hi ? 16 : 0;
    const int NIT = LL / 32;

    const char* kbase = (const char*)(Kh + (size_t)bh * LL * DPAD);
    const char* vbase = (const char*)(Vt + (size_t)bh * DPAD * LL);

    // waves 0,1 stage the K chunk (2KB contiguous); waves 2,3 stage the V chunk
    auto stage = [&](int buf, int it) {
        if (it >= NIT) return;
        const int p0 = (wave & 1) * 2;
        if (wave < 2) {
            const char* g = kbase + (size_t)it * 2048;
            char* l = (char*)&Klds[buf][0][0];
#pragma unroll
            for (int j = 0; j < 2; ++j) {
                int off = (p0 + j) * 512 + lane * 16;
                copy16(g + off, l + off);
            }
        } else {
#pragma unroll
            for (int j = 0; j < 2; ++j) {
                int dim = (p0 + j) * 8 + (lane >> 2);
                int bo  = (lane & 3) * 16;
                copy16(vbase + (size_t)dim * (LL * 2) + (size_t)it * 64 + bo,
                       (char*)&Vlds[buf][dim][0] + bo);
            }
        }
    };

    // ---- Q A-fragment (16x32 f16, contiguous k = 16*hi + s order) ----
    v16h aq;
    {
        const _Float16* qrow = Qh + ((size_t)bh * LL + q0 + l15) * DPAD;
        v8h u0 = *(const v8h*)(qrow + c0);
        v8h u1 = *(const v8h*)(qrow + c0 + 8);
#pragma unroll
        for (int j = 0; j < 8; ++j) { aq[j] = u0[j]; aq[8 + j] = u1[j]; }
    }

    v8f acc0 = {}; v8f acc1 = {};
    float psum[8];
#pragma unroll
    for (int i = 0; i < 8; ++i) psum[i] = 0.f;

    stage(0, 0);
    int cur = 0;
    for (int it = 0; it < NIT; ++it) {
        copy_fence();          // wait my async pieces of buf[cur]
        __syncthreads();       // buf[cur] complete block-wide; all past buf[cur^1] reads
        stage(cur ^ 1, it + 1);

        v16h bk0 = *(const v16h*)&Klds[cur][l15][hi * 16];
        v16h bk1 = *(const v16h*)&Klds[cur][16 + l15][hi * 16];
        v8f z = {};
        v8f s0 = __builtin_amdgcn_wmma_f32_16x16x32_f16(false, aq, false, bk0, (short)0, z, false, false);
        v8f s1 = __builtin_amdgcn_wmma_f32_16x16x32_f16(false, aq, false, bk1, (short)0, z, false, false);

#pragma unroll
        for (int i = 0; i < 8; ++i) {
            float p0 = __expf(s0[i]);
            float p1 = __expf(s1[i]);
            psum[i] += p0 + p1;
            int row = i + (hi ? 8 : 0);          // C-layout row
            p_lds[wave][row][l15]      = (_Float16)p0;
            p_lds[wave][row][16 + l15] = (_Float16)p1;
        }
        // re-read P in A-layout (same-wave DS in-order; wave-private slice)
        v16h ap;
        {
            const _Float16* prow = &p_lds[wave][l15][0];
            v8h u0 = *(const v8h*)(prow + c0);
            v8h u1 = *(const v8h*)(prow + c0 + 8);
#pragma unroll
            for (int j = 0; j < 8; ++j) { ap[j] = u0[j]; ap[8 + j] = u1[j]; }
        }
        v16h bv0 = *(const v16h*)&Vlds[cur][l15][hi * 16];
        v16h bv1 = *(const v16h*)&Vlds[cur][16 + l15][hi * 16];
        acc0 = __builtin_amdgcn_wmma_f32_16x16x32_f16(false, ap, false, bv0, (short)0, acc0, false, false);
        acc1 = __builtin_amdgcn_wmma_f32_16x16x32_f16(false, ap, false, bv1, (short)0, acc1, false, false);
        cur ^= 1;
    }

    // single cross-lane row-sum reduction + normalize + store
#pragma unroll
    for (int i = 0; i < 8; ++i) {
        float rs = psum[i];
#pragma unroll
        for (int m = 8; m >= 1; m >>= 1) rs += __shfl_xor(rs, m, 32);
        float invl = 1.0f / rs;
        int row = q0 + i + (hi ? 8 : 0);
        float* orow = O + ((size_t)bh * LL + row) * HD;
        orow[l15] = acc0[i] * invl;
        if (l15 < 8) orow[16 + l15] = acc1[i] * invl;
    }
}

// ============================================================
// Kernel 3: output projection + residual + norm2 for FFN
// ============================================================
__global__ __launch_bounds__(256) void k_outproj(
    const float* __restrict__ O, const float* __restrict__ xang, const float* __restrict__ xamp,
    const float* __restrict__ wo_amp, const float* __restrict__ bo_amp,
    const float* __restrict__ wo_ar, const float* __restrict__ wo_ai,
    const float* __restrict__ g_ang2,
    float* __restrict__ angr, float* __restrict__ angi,
    float* __restrict__ fr, float* __restrict__ fi, float* __restrict__ amp1)
{
    int t = blockIdx.x * blockDim.x + threadIdx.x;
    if (t >= BB * LL) return;
    int b = t / LL, l = t % LL;

    float mr[NH*GV], mi[NH*GV], ma[NH*AV];
#pragma unroll
    for (int h = 0; h < NH; ++h) {
        const float* oh = O + ((size_t)(b*NH + h) * LL + l) * HD;
#pragma unroll
        for (int j = 0; j < GV; ++j) { mr[h*GV+j] = oh[j]; mi[h*GV+j] = oh[GV+j]; }
#pragma unroll
        for (int j = 0; j < AV; ++j) ma[h*AV+j] = oh[2*GV+j];
    }
    const float* xm = xamp + (size_t)t * AMPC;
#pragma unroll
    for (int o = 0; o < AMPC; ++o) {
        float s = bo_amp[o];
#pragma unroll
        for (int j = 0; j < NH*AV; ++j) s += ma[j] * wo_amp[o*(NH*AV)+j];
        amp1[(size_t)t*AMPC + o] = xm[o] + s;
    }
    const float* xa = xang + (size_t)t * (2*ANGC);
    float rr[ANGC], ri[ANGC]; float ss = 0.f;
#pragma unroll
    for (int o = 0; o < ANGC; ++o) {
        float sr = 0.f, si = 0.f;
#pragma unroll
        for (int j = 0; j < NH*GV; ++j) {
            float wr = wo_ar[o*(NH*GV)+j], wi = wo_ai[o*(NH*GV)+j];
            sr += mr[j]*wr - mi[j]*wi;  si += mr[j]*wi + mi[j]*wr;
        }
        float vr = xa[o] + sr, vi = xa[ANGC+o] + si;
        rr[o] = vr; ri[o] = vi; ss += vr*vr + vi*vi;
        angr[(size_t)t*ANGC+o] = vr; angi[(size_t)t*ANGC+o] = vi;
    }
    float inv = rsqrtf(ss / ANGC + 1e-10f);
#pragma unroll
    for (int o = 0; o < ANGC; ++o) {
        float g = g_ang2[o] * inv;
        fr[(size_t)t*ANGC+o] = rr[o]*g; fi[(size_t)t*ANGC+o] = ri[o]*g;
    }
}

// ============================================================
// Kernel 4a: complex conv (K=4,S=1), out [B,128,2051]
// ============================================================
__global__ __launch_bounds__(256) void k_cconv(
    const float* __restrict__ fr, const float* __restrict__ fi,
    const float* __restrict__ wre, const float* __restrict__ wim,
    float* __restrict__ cr, float* __restrict__ ci)
{
    int idx = blockIdx.x * blockDim.x + threadIdx.x;
    if (idx >= BB * 2 * CI * TC) return;
    int tp = idx % TC; int rem = idx / TC;
    int oc = rem % (2*CI); int b = rem / (2*CI);
    float sr = 0.f, si = 0.f;
#pragma unroll
    for (int k = 0; k < KK; ++k) {
        int l = tp + k - 3;
        if (l >= 0 && l < LL) {
            const float* xr = fr + ((size_t)b*LL + l) * ANGC;
            const float* xi = fi + ((size_t)b*LL + l) * ANGC;
#pragma unroll
            for (int j = 0; j < ANGC; ++j) {
                float wr = wre[(oc*ANGC+j)*KK+k], wi = wim[(oc*ANGC+j)*KK+k];
                sr += wr*xr[j] - wi*xi[j];
                si += wr*xi[j] + wi*xr[j];
            }
        }
    }
    cr[((size_t)b*2*CI + oc)*TC + tp] = sr;
    ci[((size_t)b*2*CI + oc)*TC + tp] = si;
}

// ============================================================
// Kernel 4b: magnitude layernorm -> silu gate -> gr/gi
// ============================================================
__global__ __launch_bounds__(256) void k_cgate(
    const float* __restrict__ cr, const float* __restrict__ ci,
    const float* __restrict__ lng, const float* __restrict__ lnb,
    float* __restrict__ gr, float* __restrict__ gi)
{
    int idx = blockIdx.x * blockDim.x + threadIdx.x;
    if (idx >= BB * TC) return;
    int tp = idx % TC, b = idx / TC;
    float mag[CI]; float mu = 0.f;
#pragma unroll 8
    for (int c = 0; c < CI; ++c) {
        float a  = cr[((size_t)b*2*CI + CI + c)*TC + tp];
        float bb = ci[((size_t)b*2*CI + CI + c)*TC + tp];
        mag[c] = sqrtf(a*a + bb*bb + 1e-9f); mu += mag[c];
    }
    mu /= CI; float var = 0.f;
#pragma unroll 8
    for (int c = 0; c < CI; ++c) { float d = mag[c]-mu; var += d*d; }
    float inv = rsqrtf(var / CI + 1e-5f);
#pragma unroll 8
    for (int c = 0; c < CI; ++c) {
        float ln = (mag[c]-mu)*inv*lng[c] + lnb[c];
        float g  = ln * sigm(ln);
        gr[((size_t)b*CI + c)*TC + tp] = cr[((size_t)b*2*CI + c)*TC + tp] * g;
        gi[((size_t)b*CI + c)*TC + tp] = ci[((size_t)b*2*CI + c)*TC + tp] * g;
    }
}

// ============================================================
// Kernel 4c: complex conv-transpose + residual -> ang output (d_out[0:B*L*32])
// out[o,t] = sum_i sum_k w[i,o,k] * x[i,t-k], t = l+3 (always in range)
// ============================================================
__global__ __launch_bounds__(256) void k_cdeconv(
    const float* __restrict__ gr, const float* __restrict__ gi,
    const float* __restrict__ wre, const float* __restrict__ wim,
    const float* __restrict__ angr, const float* __restrict__ angi,
    float* __restrict__ outang)
{
    int idx = blockIdx.x * blockDim.x + threadIdx.x;
    if (idx >= BB * LL * ANGC) return;
    int oc = idx % ANGC; int rem = idx / ANGC;
    int l = rem % LL, b = rem / LL;
    int tt = l + 3;
    float sr = 0.f, si = 0.f;
    for (int i = 0; i < CI; ++i) {
        const float* xr = gr + ((size_t)b*CI + i)*TC;
        const float* xi = gi + ((size_t)b*CI + i)*TC;
#pragma unroll
        for (int k = 0; k < KK; ++k) {
            float wr = wre[(i*ANGC+oc)*KK+k], wi = wim[(i*ANGC+oc)*KK+k];
            float vr = xr[tt-k], vi = xi[tt-k];
            sr += wr*vr - wi*vi;
            si += wr*vi + wi*vr;
        }
    }
    size_t tok = (size_t)b*LL + l;
    outang[tok*(2*ANGC) + oc]        = angr[tok*ANGC+oc] + sr;
    outang[tok*(2*ANGC) + ANGC + oc] = angi[tok*ANGC+oc] + si;
}

// ============================================================
// Kernel 5: GRU input precompute xp = rmsnorm(amp)@Wih^T + bih for both dirs
// ============================================================
__global__ __launch_bounds__(256) void k_gruprep(
    const float* __restrict__ amp1, const float* __restrict__ g_amp2,
    const float* __restrict__ wf, const float* __restrict__ bf,
    const float* __restrict__ wb, const float* __restrict__ bb,
    float* __restrict__ xp)
{
    int t = blockIdx.x * blockDim.x + threadIdx.x;
    if (t >= BB * LL) return;
    float na[AMPC]; float ss = 0.f;
    const float* xm = amp1 + (size_t)t*AMPC;
#pragma unroll
    for (int j = 0; j < AMPC; ++j) { na[j] = xm[j]; ss += na[j]*na[j]; }
    float inv = rsqrtf(ss / AMPC + 1e-9f);
#pragma unroll
    for (int j = 0; j < AMPC; ++j) na[j] *= inv * g_amp2[j];
    for (int o = 0; o < 3*GH; ++o) {
        float sf = bf[o], sb = bb[o];
#pragma unroll
        for (int j = 0; j < AMPC; ++j) { sf += na[j]*wf[o*AMPC+j]; sb += na[j]*wb[o*AMPC+j]; }
        xp[(size_t)t*(3*GH) + o] = sf;
        xp[((size_t)BB*LL + t)*(3*GH) + o] = sb;
    }
}

// ============================================================
// Kernel 6: GRU sequential scan, one block per (batch,dir)
// 288 threads: thread j owns row j of h@Whh^T (Whh row cached in VGPRs)
// ============================================================
__global__ __launch_bounds__(288) void k_gruscan(
    const float* __restrict__ xp,
    const float* __restrict__ whhf, const float* __restrict__ bhhf,
    const float* __restrict__ whhb, const float* __restrict__ bhhb,
    float* __restrict__ hs)
{
    int dir = blockIdx.x & 1, b = blockIdx.x >> 1;
    const float* whh = dir ? whhb : whhf;
    const float* bhh = dir ? bhhb : bhhf;
    const float* xpb = xp + ((size_t)dir*BB*LL + (size_t)b*LL) * (3*GH);
    float* hsb       = hs + ((size_t)dir*BB*LL + (size_t)b*LL) * GH;

    __shared__ float h_lds[GH];
    __shared__ float hp_lds[3*GH];
    int j = threadIdx.x;
    float wrow[GH];
#pragma unroll
    for (int i = 0; i < GH; ++i) wrow[i] = whh[(size_t)j*GH + i];
    float bj = bhh[j];
    if (j < GH) h_lds[j] = 0.f;
    __syncthreads();

    for (int s = 0; s < LL; ++s) {
        int tt = dir ? (LL - 1 - s) : s;
        float acc = bj;
#pragma unroll
        for (int i = 0; i < GH; ++i) acc += wrow[i] * h_lds[i];
        hp_lds[j] = acc;
        __syncthreads();
        float hnew = 0.f;
        if (j < GH) {
            const float* xt = xpb + (size_t)tt * (3*GH);
            float rg = sigm(xt[j]        + hp_lds[j]);
            float zg = sigm(xt[GH + j]   + hp_lds[GH + j]);
            float ng = tanhf(xt[2*GH + j] + rg * hp_lds[2*GH + j]);
            hnew = (1.f - zg) * ng + zg * h_lds[j];
            hsb[(size_t)tt*GH + j] = hnew;
        }
        __syncthreads();
        if (j < GH) h_lds[j] = hnew;
        __syncthreads();
    }
}

// ============================================================
// Kernel 7: leaky_relu(concat) @ lin + residual -> amp output
// ============================================================
__global__ __launch_bounds__(256) void k_gruout(
    const float* __restrict__ hs, const float* __restrict__ amp1,
    const float* __restrict__ lw, const float* __restrict__ lb,
    float* __restrict__ outamp)
{
    int t = blockIdx.x * blockDim.x + threadIdx.x;
    if (t >= BB * LL) return;
    float y[2*GH];
    const float* hf = hs + (size_t)t*GH;
    const float* hb = hs + ((size_t)BB*LL + t)*GH;
#pragma unroll
    for (int j = 0; j < GH; ++j) {
        float a = hf[j]; y[j]      = a > 0.f ? a : 0.01f*a;
        float c = hb[j]; y[GH + j] = c > 0.f ? c : 0.01f*c;
    }
    for (int o = 0; o < AMPC; ++o) {
        float s = lb[o];
#pragma unroll
        for (int j = 0; j < 2*GH; ++j) s += y[j] * lw[o*(2*GH)+j];
        outamp[(size_t)t*AMPC + o] = amp1[(size_t)t*AMPC + o] + s;
    }
}

// ============================================================
// launcher
// ============================================================
extern "C" void kernel_launch(void* const* d_in, const int* in_sizes, int n_in,
                              void* d_out, int out_size, void* d_ws, size_t ws_size,
                              hipStream_t stream) {
    const float* x_ang   = (const float*)d_in[0];
    const float* x_amp   = (const float*)d_in[1];
    const float* g_ang1  = (const float*)d_in[2];
    const float* g_amp1  = (const float*)d_in[3];
    const float* wq_amp  = (const float*)d_in[4];
    const float* wk_amp  = (const float*)d_in[5];
    const float* wq_ar   = (const float*)d_in[6];
    const float* wq_ai   = (const float*)d_in[7];
    const float* wk_ar   = (const float*)d_in[8];
    const float* wk_ai   = (const float*)d_in[9];
    const float* wv_amp  = (const float*)d_in[10];
    const float* wv_ar   = (const float*)d_in[11];
    const float* wv_ai   = (const float*)d_in[12];
    const float* wo_amp  = (const float*)d_in[13];
    const float* bo_amp  = (const float*)d_in[14];
    const float* wo_ar   = (const float*)d_in[15];
    const float* wo_ai   = (const float*)d_in[16];
    const float* g_ang2  = (const float*)d_in[17];
    const float* g_amp2  = (const float*)d_in[18];
    const float* conv_re = (const float*)d_in[19];
    const float* conv_im = (const float*)d_in[20];
    const float* ln_g    = (const float*)d_in[21];
    const float* ln_b    = (const float*)d_in[22];
    const float* dconv_re= (const float*)d_in[23];
    const float* dconv_im= (const float*)d_in[24];
    const float* wih_f   = (const float*)d_in[25];
    const float* whh_f   = (const float*)d_in[26];
    const float* bih_f   = (const float*)d_in[27];
    const float* bhh_f   = (const float*)d_in[28];
    const float* wih_b   = (const float*)d_in[29];
    const float* whh_b   = (const float*)d_in[30];
    const float* bih_b   = (const float*)d_in[31];
    const float* bhh_b   = (const float*)d_in[32];
    const float* lin_w   = (const float*)d_in[33];
    const float* lin_b   = (const float*)d_in[34];

    float* out_ang = (float*)d_out;                         // [B,L,32]
    float* out_amp = out_ang + (size_t)BB*LL*(2*ANGC);      // [B,L,48]

    char* ws = (char*)d_ws;
    _Float16* Qh  = (_Float16*)(ws + 0);
    _Float16* Kh  = (_Float16*)(ws + 2097152);
    _Float16* Vt  = (_Float16*)(ws + 4194304);
    float* O      = (float*)(ws + 6291456);
    float* angr   = (float*)(ws + 9437184);
    float* angi   = (float*)(ws + 9961472);
    float* fr     = (float*)(ws + 10485760);
    float* fi     = (float*)(ws + 11010048);
    float* amp1   = (float*)(ws + 11534336);
    float* cr     = (float*)(ws + 13107200);
    float* ci     = (float*)(ws + 17307648);
    float* gr     = (float*)(ws + 21508096);
    float* gi     = (float*)(ws + 23608320);
    float* xp     = (float*)(ws + 25708544);
    float* hs     = (float*)(ws + 44582912);

    const int NT = BB * LL;                                  // 8192 tokens

    k_qkv<<<(NT + 255) / 256, 256, 0, stream>>>(
        x_ang, x_amp, g_ang1, g_amp1, wq_amp, wk_amp,
        wq_ar, wq_ai, wk_ar, wk_ai, wv_amp, wv_ar, wv_ai, Qh, Kh, Vt);

    k_flash<<<BB * NH * (LL / 64), 128, 0, stream>>>(Qh, Kh, Vt, O);

    k_outproj<<<(NT + 255) / 256, 256, 0, stream>>>(
        O, x_ang, x_amp, wo_amp, bo_amp, wo_ar, wo_ai, g_ang2,
        angr, angi, fr, fi, amp1);

    {
        int n = BB * 2 * CI * TC;
        k_cconv<<<(n + 255) / 256, 256, 0, stream>>>(fr, fi, conv_re, conv_im, cr, ci);
    }
    {
        int n = BB * TC;
        k_cgate<<<(n + 255) / 256, 256, 0, stream>>>(cr, ci, ln_g, ln_b, gr, gi);
    }
    {
        int n = BB * LL * ANGC;
        k_cdeconv<<<(n + 255) / 256, 256, 0, stream>>>(gr, gi, dconv_re, dconv_im,
                                                       angr, angi, out_ang);
    }

    k_gruprep<<<(NT + 255) / 256, 256, 0, stream>>>(
        amp1, g_amp2, wih_f, bih_f, wih_b, bih_b, xp);

    k_gruscan<<<2 * BB, 288, 0, stream>>>(xp, whh_f, bhh_f, whh_b, bhh_b, hs);

    k_gruout<<<(NT + 255) / 256, 256, 0, stream>>>(hs, amp1, lin_w, lin_b, out_amp);
}
